// DeformAttnBottleneck_35330400977531
// MI455X (gfx1250) — matrix operand order repaired and checked by reading, not compiled, not run on hardware
//
#include <hip/hip_runtime.h>
#include <hip/hip_bf16.h>

typedef __attribute__((ext_vector_type(16))) __bf16 v16bf;
typedef __attribute__((ext_vector_type(8)))  __bf16 v8bf;
typedef __attribute__((ext_vector_type(8)))  float  v8f;
typedef int vsi4 __attribute__((vector_size(16)));  // matches builtin's V4i

#define BM 128
#define BN 64
#define BK 32
#define LDPAD 40  // padded per-row LDS stride (elements): 80B, keeps 16B alignment

#if __has_builtin(__builtin_amdgcn_global_load_async_to_lds_b128)
#define HAVE_ASYNC 1
#else
#define HAVE_ASYNC 0
#endif

__device__ __forceinline__ void copy16(const __bf16* g, __bf16* l) {
#if HAVE_ASYNC
  __builtin_amdgcn_global_load_async_to_lds_b128(
      (__attribute__((address_space(1))) vsi4*)(unsigned long long)(const void*)g,
      (__attribute__((address_space(3))) vsi4*)(unsigned)(unsigned long long)(const void*)l,
      0, 0);
#else
  *(v8bf*)l = *(const v8bf*)g;
#endif
}

__device__ __forceinline__ void wait_async() {
#if HAVE_ASYNC
#if __has_builtin(__builtin_amdgcn_s_wait_asynccnt)
  __builtin_amdgcn_s_wait_asynccnt(0);
#else
  asm volatile("s_wait_asynccnt 0x0" ::: "memory");
#endif
#endif
}

__device__ __forceinline__ v8f vzero8() {
  v8f z;
#pragma unroll
  for (int i = 0; i < 8; ++i) z[i] = 0.f;
  return z;
}

// A fragment 16x32 bf16: lane (row=l&15, h=l>>4) holds K={h*8+0..7, 16+h*8+0..7}.
__device__ __forceinline__ v16bf load_a_frag(const __bf16* base, int lane) {
  int row = lane & 15, h = lane >> 4;
  const __bf16* p = base + row * LDPAD;
  v16bf a;
#pragma unroll
  for (int j = 0; j < 8; ++j) {
    a[j]     = p[h * 8 + j];
    a[8 + j] = p[16 + h * 8 + j];
  }
  return a;
}

// B fragment 32x16 bf16 from LDS [n][k]: lane (col=l&15, h=l>>4) K={h*8+0..7, 16+h*8+0..7}.
__device__ __forceinline__ v16bf load_b_frag(const __bf16* base, int lane) {
  int col = lane & 15, h = lane >> 4;
  const __bf16* p = base + col * LDPAD;
  v16bf b;
#pragma unroll
  for (int j = 0; j < 8; ++j) {
    b[j]     = p[h * 8 + j];
    b[8 + j] = p[16 + h * 8 + j];
  }
  return b;
}

// GEMM on pre-split hi/lo bf16 operands (3 WMMAs/tile ~ fp32 accuracy).
// A: row-major hi/lo bf16 (lda), or fp32 conv layout x[b,k,hw] if Aconv!=null.
// B: pre-transposed bf16 Bt[n][k] (padded to 64-row multiple, zero filled).
// epi: 0 +bias, 1 +bias,ReLU, 2 BN+ReLU, 3 BN+ReLU scatter to (B,N,4096).
// Optional fp32 C and/or hi/lo bf16 CHi/CLo outputs (row-major ldc).
__global__ __launch_bounds__(256) void gemm_kernel(
    const __bf16* __restrict__ Ahi, const __bf16* __restrict__ Alo,
    const float* __restrict__ Aconv,
    const __bf16* __restrict__ Bth, const __bf16* __restrict__ Btl,
    float* __restrict__ C, __bf16* __restrict__ CHi, __bf16* __restrict__ CLo,
    int M, int N, int K, int lda, int ldc,
    const float* __restrict__ p0, const float* __restrict__ p1,
    const float* __restrict__ p2, const float* __restrict__ p3, int epi)
{
  __shared__ __align__(16) __bf16 lAhi[BM * LDPAD];
  __shared__ __align__(16) __bf16 lAlo[BM * LDPAD];
  __shared__ __align__(16) __bf16 lBhi[BN * LDPAD];
  __shared__ __align__(16) __bf16 lBlo[BN * LDPAD];

  const int tid  = threadIdx.x;
  const int lane = tid & 31;
  const int wv   = tid >> 5;
  const int wr   = wv >> 1;
  const int wc   = wv & 1;
  const int bm0  = blockIdx.x * BM;
  const int bn0  = blockIdx.y * BN;

  v8f acc[2][2];
#pragma unroll
  for (int i = 0; i < 2; ++i)
#pragma unroll
    for (int j = 0; j < 2; ++j) acc[i][j] = vzero8();

  for (int kk = 0; kk < K; kk += BK) {
    __syncthreads();
    if (!Aconv) {
      const __bf16* Ah = Ahi + bm0 * lda + kk;
      const __bf16* Al = Alo + bm0 * lda + kk;
#pragma unroll
      for (int it = 0; it < 4; ++it) {  // 2 bufs x 128 rows x 4 k-quads
        int s   = tid + it * 256;
        int buf = s >> 9;
        int r   = (s >> 2) & 127;
        int kq  = (s & 3) * 8;
        copy16((buf ? Al : Ah) + r * lda + kq,
               (buf ? lAlo : lAhi) + r * LDPAD + kq);
      }
    } else {  // proj_in: fp32 x[b,k,hw], vectorized over hw, split on the fly
      const float* aB = Aconv + (bm0 >> 12) * K * 4096 + (bm0 & 4095);
#pragma unroll
      for (int it = 0; it < 4; ++it) {  // 32 m-quads x 32 k
        int s  = tid + it * 256;
        int m4 = (s & 31) * 4;
        int k  = s >> 5;
        float4 f4 = *(const float4*)(aB + (kk + k) * 4096 + m4);
        float fv[4] = {f4.x, f4.y, f4.z, f4.w};
#pragma unroll
        for (int j = 0; j < 4; ++j) {
          __bf16 hi = (__bf16)fv[j];
          lAhi[(m4 + j) * LDPAD + k] = hi;
          lAlo[(m4 + j) * LDPAD + k] = (__bf16)(fv[j] - (float)hi);
        }
      }
    }
    {
      const __bf16* Bh = Bth + bn0 * K + kk;
      const __bf16* Bl = Btl + bn0 * K + kk;
#pragma unroll
      for (int it = 0; it < 2; ++it) {  // 2 bufs x 64 cols x 4 k-quads
        int s   = tid + it * 256;
        int buf = s >> 8;
        int n   = (s >> 2) & 63;
        int kq  = (s & 3) * 8;
        copy16((buf ? Bl : Bh) + n * K + kq,
               (buf ? lBlo : lBhi) + n * LDPAD + kq);
      }
    }
    wait_async();
    __syncthreads();

#pragma unroll
    for (int ti = 0; ti < 2; ++ti) {
      v16bf ahi = load_a_frag(&lAhi[(wr * 32 + ti * 16) * LDPAD], lane);
      v16bf alo = load_a_frag(&lAlo[(wr * 32 + ti * 16) * LDPAD], lane);
#pragma unroll
      for (int tj = 0; tj < 2; ++tj) {
        v16bf bhi = load_b_frag(&lBhi[(wc * 32 + tj * 16) * LDPAD], lane);
        v16bf blo = load_b_frag(&lBlo[(wc * 32 + tj * 16) * LDPAD], lane);
        acc[ti][tj] = __builtin_amdgcn_wmma_f32_16x16x32_bf16(
            false, ahi, false, bhi, (short)0, acc[ti][tj], false, false);
        acc[ti][tj] = __builtin_amdgcn_wmma_f32_16x16x32_bf16(
            false, ahi, false, blo, (short)0, acc[ti][tj], false, false);
        acc[ti][tj] = __builtin_amdgcn_wmma_f32_16x16x32_bf16(
            false, alo, false, bhi, (short)0, acc[ti][tj], false, false);
      }
    }
  }

#pragma unroll
  for (int ti = 0; ti < 2; ++ti) {
#pragma unroll
    for (int tj = 0; tj < 2; ++tj) {
      int nn = bn0 + wc * 32 + tj * 16 + (lane & 15);
      if (nn >= N) continue;
      int mB = bm0 + wr * 32 + ti * 16 + 8 * (lane >> 4);
#pragma unroll
      for (int r = 0; r < 8; ++r) {
        int m   = mB + r;
        float v = acc[ti][tj][r];
        float t;
        int ci = m * ldc + nn;
        if (epi == 0) {
          t = v + p0[nn];
        } else if (epi == 1) {
          t = v + p0[nn];
          t = t > 0.f ? t : 0.f;
        } else {
          t = (v - p2[nn]) * rsqrtf(p3[nn] + 1e-5f) * p0[nn] + p1[nn];
          t = t > 0.f ? t : 0.f;
          if (epi == 3) ci = ((m >> 12) * N + nn) * 4096 + (m & 4095);
        }
        if (C) C[ci] = t;
        if (CHi) {
          __bf16 hi = (__bf16)t;
          CHi[ci] = hi;
          CLo[ci] = (__bf16)(t - (float)hi);
        }
      }
    }
  }
}

// Weight pre-transpose + hi/lo split: out[n*K+k] from W[k*sbk + n*sbn],
// padded/zeroed to Npad rows. K is a power of two (kshift).
__global__ __launch_bounds__(256) void convw_kernel(
    const float* __restrict__ W, int sbk, int sbn, int kshift, int N,
    __bf16* __restrict__ oh, __bf16* __restrict__ ol)
{
  int idx = blockIdx.x * 256 + threadIdx.x;
  int n = idx >> kshift;
  int k = idx & ((1 << kshift) - 1);
  float f = (n < N) ? W[k * sbk + n * sbn] : 0.f;
  __bf16 hi = (__bf16)f;
  oh[idx] = hi;
  ol[idx] = (__bf16)(f - (float)hi);
}

// q init: broadcast query_embed over batch, hi/lo split.
__global__ __launch_bounds__(256) void init_q_kernel(
    __bf16* __restrict__ qh, __bf16* __restrict__ ql, const float* __restrict__ qe)
{
  int i = blockIdx.x * 256 + threadIdx.x;
  float f = qe[i & (4096 * 256 - 1)];
  __bf16 hi = (__bf16)f;
  qh[i] = hi;
  ql[i] = (__bf16)(f - (float)hi);
}

__device__ __forceinline__ float corner(const float* vb, int xi, int yi) {
  if (xi < 0 || xi > 63 || yi < 0 || yi > 63) return 0.f;
  return vb[(yi * 64 + xi) * 256];
}

// Deformable sampling: one wave per (b,query,head); lanes = 32 head channels.
__global__ __launch_bounds__(256) void samp_kernel(
    const float* __restrict__ val, const float* __restrict__ off,
    const float* __restrict__ logit,
    __bf16* __restrict__ oh, __bf16* __restrict__ ol)
{
  int gw   = blockIdx.x * 8 + (threadIdx.x >> 5);
  int lane = threadIdx.x & 31;
  int head = gw & 7;
  int m    = gw >> 3;
  int b    = m >> 12, hw = m & 4095;
  int qx   = hw & 63,  qy = hw >> 6;

  const float* lg = logit + m * 32 + head * 4;
  float l0 = lg[0], l1 = lg[1], l2 = lg[2], l3 = lg[3];
  float mx = fmaxf(fmaxf(l0, l1), fmaxf(l2, l3));
  float e0 = expf(l0 - mx), e1 = expf(l1 - mx), e2 = expf(l2 - mx), e3 = expf(l3 - mx);
  float inv = 1.f / (e0 + e1 + e2 + e3);
  float aw[4] = {e0 * inv, e1 * inv, e2 * inv, e3 * inv};

  const float* of = off + m * 64 + head * 8;
  const float* vb = val + b * 4096 * 256 + head * 32 + lane;

  float acc = 0.f;
#pragma unroll
  for (int p = 0; p < 4; ++p) {
    float gx = (float)qx + of[p * 2 + 0];
    float gy = (float)qy + of[p * 2 + 1];
    float x0f = floorf(gx), y0f = floorf(gy);
    float wx = gx - x0f, wy = gy - y0f;
    int x0 = (int)x0f, y0 = (int)y0f;
    float ap = aw[p];
    acc += ap * (1.f - wx) * (1.f - wy) * corner(vb, x0,     y0);
    acc += ap * wx         * (1.f - wy) * corner(vb, x0 + 1, y0);
    acc += ap * (1.f - wx) * wy         * corner(vb, x0,     y0 + 1);
    acc += ap * wx         * wy         * corner(vb, x0 + 1, y0 + 1);
  }
  int ci = m * 256 + head * 32 + lane;
  __bf16 hi = (__bf16)acc;
  oh[ci] = hi;
  ol[ci] = (__bf16)(acc - (float)hi);
}

// q = LayerNorm((qhi+qlo) + res); one wave per token; writes hi/lo back.
__global__ __launch_bounds__(256) void ln_add_kernel(
    __bf16* __restrict__ qh, __bf16* __restrict__ ql,
    const float* __restrict__ res,
    const float* __restrict__ g, const float* __restrict__ b)
{
  int m    = blockIdx.x * 8 + (threadIdx.x >> 5);
  int lane = threadIdx.x & 31;
  float x[8];
  float s = 0.f;
#pragma unroll
  for (int j = 0; j < 8; ++j) {
    int i = m * 256 + j * 32 + lane;
    x[j] = (float)qh[i] + (float)ql[i] + res[i];
    s += x[j];
  }
#pragma unroll
  for (int o = 16; o; o >>= 1) s += __shfl_xor(s, o, 32);
  float mean = s * (1.f / 256.f);
  float v = 0.f;
#pragma unroll
  for (int j = 0; j < 8; ++j) { float d = x[j] - mean; v += d * d; }
#pragma unroll
  for (int o = 16; o; o >>= 1) v += __shfl_xor(v, o, 32);
  float rs = rsqrtf(v * (1.f / 256.f) + 1e-5f);
#pragma unroll
  for (int j = 0; j < 8; ++j) {
    int d = j * 32 + lane;
    int i = m * 256 + d;
    float t = (x[j] - mean) * rs * g[d] + b[d];
    __bf16 hi = (__bf16)t;
    qh[i] = hi;
    ql[i] = (__bf16)(t - (float)hi);
  }
}

static void gemm(hipStream_t s, const __bf16* Ah, const __bf16* Al,
                 const float* Aconv, const __bf16* Bh, const __bf16* Bl,
                 float* C, __bf16* CHi, __bf16* CLo,
                 int M, int N, int K, int lda, int ldc,
                 const float* p0, const float* p1, const float* p2,
                 const float* p3, int epi)
{
  dim3 grid(M / BM, (N + BN - 1) / BN);
  gemm_kernel<<<grid, 256, 0, s>>>(Ah, Al, Aconv, Bh, Bl, C, CHi, CLo,
                                   M, N, K, lda, ldc, p0, p1, p2, p3, epi);
}

extern "C" void kernel_launch(void* const* d_in, const int* in_sizes, int n_in,
                              void* d_out, int out_size, void* d_ws, size_t ws_size,
                              hipStream_t stream) {
  (void)in_sizes; (void)n_in; (void)out_size; (void)ws_size;
  const int L = 2, D = 256, DFF = 1024, CIN = 512;
  const int M = 8 * 4096;
  const long MD = (long)M * D;

  const float* x    = (const float*)d_in[0];
  const float* Win  = (const float*)d_in[1];
  const float* bn1g = (const float*)d_in[2];
  const float* bn1b = (const float*)d_in[3];
  const float* bn1m = (const float*)d_in[4];
  const float* bn1v = (const float*)d_in[5];
  const float* qe   = (const float*)d_in[6];
  const float* Woff = (const float*)d_in[7];
  const float* boff = (const float*)d_in[8];
  const float* Wattn= (const float*)d_in[9];
  const float* battn= (const float*)d_in[10];
  const float* Wval = (const float*)d_in[11];
  const float* bval = (const float*)d_in[12];
  const float* Wo   = (const float*)d_in[13];
  const float* bo   = (const float*)d_in[14];
  const float* ln1g = (const float*)d_in[15];
  const float* ln1b = (const float*)d_in[16];
  const float* W1   = (const float*)d_in[17];
  const float* b1   = (const float*)d_in[18];
  const float* W2   = (const float*)d_in[19];
  const float* b2   = (const float*)d_in[20];
  const float* ln2g = (const float*)d_in[21];
  const float* ln2b = (const float*)d_in[22];
  const float* Wout = (const float*)d_in[23];
  const float* bn2g = (const float*)d_in[24];
  const float* bn2b = (const float*)d_in[25];
  const float* bn2m = (const float*)d_in[26];
  const float* bn2v = (const float*)d_in[27];

  const int CH = 2048, NCH = M / CH;

  // fp32 scratch
  float* tmp  = (float*)d_ws;        // M x D
  float* val  = tmp + MD;            // M x D
  float* offb = val + MD;            // M x 64
  float* atl  = offb + (long)M * 64; // M x 32
  // bf16 scratch (hi/lo pairs, lo = hi + size)
  __bf16* cur = (__bf16*)(atl + (long)M * 32);
  auto alloc = [&](long n) { __bf16* p = cur; cur += 2 * n; return p; };
  __bf16* srcH = alloc(MD);                // layer input (proj_in out)
  __bf16* qH   = alloc(MD);                // evolving queries
  __bf16* aoH  = alloc(MD);                // attention output
  __bf16* hidH = alloc((long)CH * DFF);    // FFN hidden chunk
  __bf16* winT = alloc((long)D * CIN);     // 256 x 512
  __bf16* woutT= alloc((long)CIN * D);     // 512 x 256
  __bf16 *wvalT[2], *woffT[2], *wattnT[2], *woT[2], *w1T[2], *w2T[2];
  for (int i = 0; i < L; ++i) {
    wvalT[i] = alloc((long)D * D);
    woffT[i] = alloc(64L * D);
    wattnT[i]= alloc(64L * D);   // padded from 32 to 64 rows
    woT[i]   = alloc((long)D * D);
    w1T[i]   = alloc((long)DFF * D);
    w2T[i]   = alloc((long)D * DFF);
  }
  const long sA = MD, sHid = (long)CH * DFF;

  // weight pre-transpose + split (Bt[n][k], zero-padded rows)
  auto convw = [&](const float* W, int sbk, int sbn, int kshift, int N, int Npad,
                   __bf16* o, long sz) {
    convw_kernel<<<(Npad << kshift) / 256, 256, 0, stream>>>(
        W, sbk, sbn, kshift, N, o, o + sz);
  };
  convw(Win, 1, CIN, 9, D, D, winT, (long)D * CIN);
  convw(Wout, 1, D, 8, CIN, CIN, woutT, (long)CIN * D);
  for (int i = 0; i < L; ++i) {
    convw(Wval + (long)i * D * D,    D,   1, 8,  D,   D,   wvalT[i], (long)D * D);
    convw(Woff + (long)i * D * 64,   64,  1, 8,  64,  64,  woffT[i], 64L * D);
    convw(Wattn + (long)i * D * 32,  32,  1, 8,  32,  64,  wattnT[i], 64L * D);
    convw(Wo + (long)i * D * D,      D,   1, 8,  D,   D,   woT[i],   (long)D * D);
    convw(W1 + (long)i * D * DFF,    DFF, 1, 8,  DFF, DFF, w1T[i],   (long)DFF * D);
    convw(W2 + (long)i * DFF * D,    D,   1, 10, D,   D,   w2T[i],   (long)D * DFF);
  }

  // proj_in: conv1x1 + BN + ReLU -> srcH/L (bf16 split only)
  gemm(stream, nullptr, nullptr, x, winT, winT + (long)D * CIN,
       nullptr, srcH, srcH + sA, M, D, CIN, 0, D, bn1g, bn1b, bn1m, bn1v, 2);
  init_q_kernel<<<(int)(MD / 256), 256, 0, stream>>>(qH, qH + sA, qe);

  for (int i = 0; i < L; ++i) {
    gemm(stream, srcH, srcH + sA, nullptr, wvalT[i], wvalT[i] + (long)D * D,
         val, nullptr, nullptr, M, D, D, D, D,
         bval + (long)i * D, nullptr, nullptr, nullptr, 0);
    gemm(stream, qH, qH + sA, nullptr, woffT[i], woffT[i] + 64L * D,
         offb, nullptr, nullptr, M, 64, D, D, 64,
         boff + (long)i * 64, nullptr, nullptr, nullptr, 0);
    gemm(stream, qH, qH + sA, nullptr, wattnT[i], wattnT[i] + 64L * D,
         atl, nullptr, nullptr, M, 32, D, D, 32,
         battn + (long)i * 32, nullptr, nullptr, nullptr, 0);
    samp_kernel<<<M, 256, 0, stream>>>(val, offb, atl, aoH, aoH + sA);
    gemm(stream, aoH, aoH + sA, nullptr, woT[i], woT[i] + (long)D * D,
         tmp, nullptr, nullptr, M, D, D, D, D,
         bo + (long)i * D, nullptr, nullptr, nullptr, 0);
    ln_add_kernel<<<M / 8, 256, 0, stream>>>(qH, qH + sA, tmp,
        ln1g + (long)i * D, ln1b + (long)i * D);
    for (int c = 0; c < NCH; ++c) {
      gemm(stream, qH + (long)c * CH * D, qH + sA + (long)c * CH * D, nullptr,
           w1T[i], w1T[i] + (long)DFF * D, nullptr, hidH, hidH + sHid,
           CH, DFF, D, D, DFF, b1 + (long)i * DFF, nullptr, nullptr, nullptr, 1);
      gemm(stream, hidH, hidH + sHid, nullptr, w2T[i], w2T[i] + (long)D * DFF,
           tmp + (long)c * CH * D, nullptr, nullptr, CH, D, DFF, DFF, D,
           b2 + (long)i * D, nullptr, nullptr, nullptr, 0);
    }
    ln_add_kernel<<<M / 8, 256, 0, stream>>>(qH, qH + sA, tmp,
        ln2g + (long)i * D, ln2b + (long)i * D);
  }

  // proj_out: (M,D)@(D,CIN) -> BN+ReLU, scattered to (B, CIN, H, W)
  gemm(stream, qH, qH + sA, nullptr, woutT, woutT + (long)CIN * D,
       (float*)d_out, nullptr, nullptr, M, CIN, D, D, 0,
       bn2g, bn2b, bn2m, bn2v, 3);
}
